// Physics_Attention_Irregular_Mesh_70815420776669
// MI455X (gfx1250) — matrix operand order repaired and checked by reading, not compiled, not run on hardware
//
#include <hip/hip_runtime.h>

// ---------------------------------------------------------------------------
// Problem constants
// ---------------------------------------------------------------------------
#define DIMX   256
#define HEADS  8
#define DHEAD  64
#define GSL    64
#define INNER  512          // HEADS*DHEAD
#define NTOK   32768        // tokens per batch
#define BATCH  4
#define TM     32           // tokens per block tile
#define TILES  (NTOK / TM)  // 1024 tiles per batch

// LDS strides (ushort elements) -- all chosen so row stride is 16B-aligned
#define XS_LD  264          // x tile row stride (256 + pad), 528 B rows
#define WL_LD  72           // per-head [32][72] slice-weight tiles, 144 B rows
#define TMP_LD 520          // [32][520] bf16 intermediate, 1040 B rows

typedef __attribute__((ext_vector_type(16))) __bf16 v16bf;
typedef __attribute__((ext_vector_type(8)))  __bf16 v8bf;
typedef __attribute__((ext_vector_type(8)))  float  v8f;

// ---------------------------------------------------------------------------
// Small helpers
// ---------------------------------------------------------------------------
__device__ __forceinline__ unsigned short f2bf(float f) {
    unsigned int u = __builtin_bit_cast(unsigned int, f);
    u += 0x7fffu + ((u >> 16) & 1u);          // round-to-nearest-even
    return (unsigned short)(u >> 16);
}
__device__ __forceinline__ unsigned int f2bf2(float lo, float hi) {
    return (unsigned int)f2bf(lo) | ((unsigned int)f2bf(hi) << 16);
}
__device__ __forceinline__ __bf16 bcu(unsigned short s) {
    return __builtin_bit_cast(__bf16, s);
}
__device__ __forceinline__ v8f vzero8() {
    v8f z;
#pragma unroll
    for (int i = 0; i < 8; ++i) z[i] = 0.0f;
    return z;
}
__device__ __forceinline__ v8f wmma_bf16(v16bf a, v16bf b, v8f c) {
    return __builtin_amdgcn_wmma_f32_16x16x32_bf16(false, a, false, b,
                                                   (short)0, c, false, false);
}

// B-fragment swizzle index for a row-major [K][N] matrix, K tiled by 32:
// lane (n, half) of K-block kk consumes K = kk*32 + half*8 + {0..7, 16..23} at fixed n,
// stored contiguously. swz[((k>>5)*N + n)*32 + (k&31)] = W[k][n].
__host__ __device__ __forceinline__ size_t swzB(int k, int n, int N) {
    return ((size_t)(k >> 5) * N + n) * 32 + (k & 31);
}

// A fragment (16xK tile, row-major bf16 source, 16B-aligned rows).
// lane holds K runs [kbase+h8, +8) and [kbase+16+h8, +8) of row m.
__device__ __forceinline__ v16bf ldA(const unsigned short* p, int ldu,
                                     int row0, int kbase, int lane) {
    const int m = row0 + (lane & 15);
    const int h8 = (lane >> 4) * 8;
    const unsigned short* q = p + (size_t)m * ldu + kbase + h8;
    v8bf lo = *(const v8bf*)q;
    v8bf hi = *(const v8bf*)(q + 16);
    return __builtin_shufflevector(lo, hi, 0, 1, 2, 3, 4, 5, 6, 7,
                                   8, 9, 10, 11, 12, 13, 14, 15);
}

// Transposed A fragment: logical A[m][k] = mem[k][m] (mem row-major [K][M]).
__device__ __forceinline__ v16bf ldAT(const unsigned short* p, int ldu,
                                      int m0, int kbase, int lane) {
    const int m = m0 + (lane & 15);
    const int h8 = (lane >> 4) * 8;
    v16bf a;
#pragma unroll
    for (int i = 0; i < 8; ++i) {
        int k = kbase + ((i & 4) ? 16 : 0) + h8 + (i & 3) * 2;
        a[2 * i]     = bcu(p[(size_t)k * ldu + m]);
        a[2 * i + 1] = bcu(p[(size_t)(k + 1) * ldu + m]);
    }
    return a;
}

// B fragment from pre-swizzled storage: two 16-byte loads, K runs [h8, +8) and [16+h8, +8).
__device__ __forceinline__ v16bf ldBswz(const unsigned short* p, int N,
                                        int kk, int n0, int lane) {
    const int n = n0 + (lane & 15);
    const unsigned short* q = p + ((size_t)(kk * N + n)) * 32 + (lane >> 4) * 8;
    v8bf lo = *(const v8bf*)q;
    v8bf hi = *(const v8bf*)(q + 16);
    return __builtin_shufflevector(lo, hi, 0, 1, 2, 3, 4, 5, 6, 7,
                                   8, 9, 10, 11, 12, 13, 14, 15);
}

// ---------------------------------------------------------------------------
// Shared device routine: slice-weight computation for one head (one wave).
// logits = (xs @ W_xs[:, h*64:+64] + bcomb) * invT + 0.1*sb ; softmax over g.
// Writes bf16 weights [32 tok][WL_LD] to wlsh; optionally accumulates per-g
// column sums into normAcc[64] with f32 atomics.
// ---------------------------------------------------------------------------
__device__ __forceinline__ void compute_slice_weights(
    const unsigned short* __restrict__ xs,    // LDS [TM][XS_LD] bf16
    const float*          __restrict__ sbls,  // LDS [TM][64] f32
    const unsigned short* __restrict__ Wxs,   // global, B-swizzled [256x512]
    const float*          __restrict__ bcomb, // global [512] f32
    float invT, unsigned short* __restrict__ wlsh, // LDS [TM][WL_LD]
    int h, int lane, float* __restrict__ normAcc)  // normAcc may be null
{
    v8f acc[2][4];
#pragma unroll
    for (int r = 0; r < 2; ++r)
#pragma unroll
        for (int c = 0; c < 4; ++c) acc[r][c] = vzero8();

#pragma unroll
    for (int kk = 0; kk < 8; ++kk) {
        v16bf a0 = ldA(xs, XS_LD, 0,  kk * 32, lane);
        v16bf a1 = ldA(xs, XS_LD, 16, kk * 32, lane);
#pragma unroll
        for (int c = 0; c < 4; ++c) {
            v16bf b = ldBswz(Wxs, INNER, kk, h * 64 + 16 * c, lane);
            acc[0][c] = wmma_bf16(a0, b, acc[0][c]);
            acc[1][c] = wmma_bf16(a1, b, acc[1][c]);
        }
    }

    const int n15 = lane & 15;
    const int half = lane >> 4;

    // bias + temperature + spatial bias
#pragma unroll
    for (int r = 0; r < 2; ++r)
#pragma unroll
        for (int c = 0; c < 4; ++c) {
            float bc = bcomb[h * 64 + 16 * c + n15];
#pragma unroll
            for (int v = 0; v < 8; ++v) {
                int tok = 16 * r + v + 8 * half;
                acc[r][c][v] = (acc[r][c][v] + bc) * invT
                             + 0.1f * sbls[tok * GSL + 16 * c + n15];
            }
        }

    // softmax across g (64 cols = 4 tiles x 16 lanes of this half)
#pragma unroll
    for (int r = 0; r < 2; ++r)
#pragma unroll
        for (int v = 0; v < 8; ++v) {
            float mx = acc[r][0][v];
#pragma unroll
            for (int c = 1; c < 4; ++c) mx = fmaxf(mx, acc[r][c][v]);
            mx = fmaxf(mx, __shfl_xor(mx, 1, 32));
            mx = fmaxf(mx, __shfl_xor(mx, 2, 32));
            mx = fmaxf(mx, __shfl_xor(mx, 4, 32));
            mx = fmaxf(mx, __shfl_xor(mx, 8, 32));
            float e[4], s = 0.0f;
#pragma unroll
            for (int c = 0; c < 4; ++c) { e[c] = __expf(acc[r][c][v] - mx); s += e[c]; }
            s += __shfl_xor(s, 1, 32);
            s += __shfl_xor(s, 2, 32);
            s += __shfl_xor(s, 4, 32);
            s += __shfl_xor(s, 8, 32);
            float inv = __builtin_amdgcn_rcpf(s);
#pragma unroll
            for (int c = 0; c < 4; ++c) acc[r][c][v] = e[c] * inv;
        }

    // store bf16 weights [tok][g] to LDS
#pragma unroll
    for (int r = 0; r < 2; ++r)
#pragma unroll
        for (int c = 0; c < 4; ++c)
#pragma unroll
            for (int v = 0; v < 8; ++v) {
                int tok = 16 * r + v + 8 * half;
                wlsh[tok * WL_LD + 16 * c + n15] = f2bf(acc[r][c][v]);
            }

    if (normAcc) {
#pragma unroll
        for (int c = 0; c < 4; ++c) {
            float p = 0.0f;
#pragma unroll
            for (int r = 0; r < 2; ++r)
#pragma unroll
                for (int v = 0; v < 8; ++v) p += acc[r][c][v];
            p += __shfl_xor(p, 16, 32);
            if (lane < 16) atomicAdd(&normAcc[16 * c + lane], p);
        }
    }
}

// ---------------------------------------------------------------------------
// Kernel 0: weight prep. B-swizzled bf16 W_fx / W_out, combined
// W_xs = W_x @ blockdiag(W_slice) (B-swizzled bf16), bcomb = b_x@W_slice+b_sl.
// ---------------------------------------------------------------------------
__global__ __launch_bounds__(256) void prep_kernel(
    const float* __restrict__ W_fx, const float* __restrict__ W_x,
    const float* __restrict__ W_slice, const float* __restrict__ b_x,
    const float* __restrict__ b_slice, const float* __restrict__ W_out,
    unsigned short* __restrict__ WfxSwz, unsigned short* __restrict__ WxsSwz,
    unsigned short* __restrict__ WoutSwz, float* __restrict__ bcomb)
{
    int i = blockIdx.x * blockDim.x + threadIdx.x;   // 0 .. 131071
    if (i < DIMX * INNER) {
        {   // W_fx: [K=256][N=512]
            int k = i >> 9, n = i & 511;
            WfxSwz[swzB(k, n, INNER)] = f2bf(W_fx[i]);
        }
        {   // W_out: [K=512][N=256]
            int k = i >> 8, n = i & 255;
            WoutSwz[swzB(k, n, DIMX)] = f2bf(W_out[i]);
        }
        {   // combined W_xs: [K=256][N=512]
            int k = i >> 9, n = i & 511;
            int h = n >> 6, g = n & 63;
            float s = 0.0f;
            for (int c = 0; c < DHEAD; ++c)
                s += W_x[k * INNER + h * 64 + c] * W_slice[c * GSL + g];
            WxsSwz[swzB(k, n, INNER)] = f2bf(s);
        }
    }
    if (i < INNER) {
        int h = i >> 6, g = i & 63;
        float s = 0.0f;
        for (int c = 0; c < DHEAD; ++c)
            s += b_x[h * 64 + c] * W_slice[c * GSL + g];
        bcomb[i] = s + b_slice[g];
    }
}

__global__ __launch_bounds__(256) void zero_kernel(float* __restrict__ p, int n)
{
    int i = blockIdx.x * blockDim.x + threadIdx.x;
    if (i < n) p[i] = 0.0f;
}

// ---------------------------------------------------------------------------
// Phase A: per 32-token tile, slice weights (fused GEMM+softmax), fx
// projection, slice_token/slice_norm accumulation via f32 atomics.
// 8 waves = 8 heads.
// ---------------------------------------------------------------------------
__global__ __launch_bounds__(256) void phaseA_kernel(
    const float* __restrict__ x, const float* __restrict__ sb,
    const unsigned short* __restrict__ WxsSwz,
    const unsigned short* __restrict__ WfxSwz,
    const float* __restrict__ bcomb, const float* __restrict__ bfx,
    const float* __restrict__ temp,
    float* __restrict__ stAcc, float* __restrict__ normAcc)
{
    extern __shared__ unsigned char smem[];
    unsigned short* xs   = (unsigned short*)smem;                    // [TM][XS_LD]
    float*          sbls = (float*)(smem + TM * XS_LD * 2);          // [TM][64]
    unsigned short* wls  = (unsigned short*)(sbls + TM * GSL);       // [8][TM][WL_LD]
    unsigned short* fxcm = wls + HEADS * TM * WL_LD;                 // [8][64 c][32 tok]

    const int tid  = threadIdx.x;
    const int wave = tid >> 5;
    const int lane = tid & 31;
    const int batch = blockIdx.x / TILES;
    const int tile  = blockIdx.x % TILES;
    const int tok0  = tile * TM;

    if (tid == 0) { __builtin_prefetch(WxsSwz, 0, 1); __builtin_prefetch(WfxSwz, 0, 1); }

    // cooperative load: x tile -> bf16 LDS (packed u32 stores), sb tile -> f32
    const float* xg = x + ((size_t)batch * NTOK + tok0) * DIMX;
    for (int i = tid; i < TM * DIMX / 4; i += 256) {
        float4 f = ((const float4*)xg)[i];
        int e = i * 4, row = e >> 8, col = e & 255;
        unsigned int* d = (unsigned int*)(xs + row * XS_LD + col);
        d[0] = f2bf2(f.x, f.y);
        d[1] = f2bf2(f.z, f.w);
    }
    const float* sbg = sb + ((size_t)batch * NTOK + tok0) * GSL;
    for (int i = tid; i < TM * GSL; i += 256) sbls[i] = sbg[i];
    __syncthreads();

    const int h = wave;
    const float invT = __builtin_amdgcn_rcpf(temp[h]);
    unsigned short* wlsh = wls + h * TM * WL_LD;
    unsigned short* fxh  = fxcm + h * DHEAD * TM;   // column-major [c][tok]

    compute_slice_weights(xs, sbls, WxsSwz, bcomb, invT, wlsh, h, lane,
                          normAcc + ((size_t)batch * HEADS + h) * GSL);

    // fx projection for this head: [32 tok][64 c], stored column-major (=B swz)
    {
        v8f acc[2][4];
#pragma unroll
        for (int r = 0; r < 2; ++r)
#pragma unroll
            for (int c = 0; c < 4; ++c) acc[r][c] = vzero8();
#pragma unroll
        for (int kk = 0; kk < 8; ++kk) {
            v16bf a0 = ldA(xs, XS_LD, 0,  kk * 32, lane);
            v16bf a1 = ldA(xs, XS_LD, 16, kk * 32, lane);
#pragma unroll
            for (int c = 0; c < 4; ++c) {
                v16bf b = ldBswz(WfxSwz, INNER, kk, h * 64 + 16 * c, lane);
                acc[0][c] = wmma_bf16(a0, b, acc[0][c]);
                acc[1][c] = wmma_bf16(a1, b, acc[1][c]);
            }
        }
        const int n15 = lane & 15, half = lane >> 4;
#pragma unroll
        for (int r = 0; r < 2; ++r)
#pragma unroll
            for (int c = 0; c < 4; ++c) {
                float bb = bfx[h * 64 + 16 * c + n15];
#pragma unroll
                for (int v = 0; v < 8; ++v) {
                    int tok = 16 * r + v + 8 * half;
                    fxh[(16 * c + n15) * TM + tok] = f2bf(acc[r][c][v] + bb);
                }
            }
    }

    // slice_token partial: st[g][c] += w^T @ fx  (K = 32 tokens, one WMMA/tile)
    {
        float* stb = stAcc + ((size_t)batch * HEADS + h) * GSL * DHEAD;
        const int n15 = lane & 15, half = lane >> 4;
#pragma unroll
        for (int rt = 0; rt < 4; ++rt) {
            v16bf a = ldAT(wlsh, WL_LD, 16 * rt, 0, lane);  // A[g][tok]
#pragma unroll
            for (int ct = 0; ct < 4; ++ct) {
                v16bf b = ldBswz(fxh, TM * 2, 0, 16 * ct, lane); // [c][tok] == swz
                v8f o = vzero8();
                o = wmma_bf16(a, b, o);
#pragma unroll
                for (int v = 0; v < 8; ++v) {
                    int g = 16 * rt + v + 8 * half;
                    atomicAdd(&stb[g * DHEAD + 16 * ct + n15], o[v]);
                }
            }
        }
    }
}

// ---------------------------------------------------------------------------
// Phase B: tiny cross-attention over slice tokens. One block per batch.
// Writes out_slice directly in B-swizzled bf16 layout for phase C.
// ---------------------------------------------------------------------------
__global__ __launch_bounds__(256) void phaseB_kernel(
    const float* __restrict__ stAcc, const float* __restrict__ normAcc,
    const float* __restrict__ Wq, const float* __restrict__ Wk,
    const float* __restrict__ Wv, const float* __restrict__ srs_p,
    const float* __restrict__ attn_scale, unsigned short* __restrict__ osSwz)
{
    __shared__ float kv[GSL * DHEAD];
    __shared__ float kn[GSL * DHEAD];
    __shared__ float vv[GSL * DHEAD];
    __shared__ float rn[GSL];

    const int b = blockIdx.x;
    const int tid = threadIdx.x;
    const float* st  = stAcc   + (size_t)b * HEADS * GSL * DHEAD;
    const float* nrm = normAcc + (size_t)b * HEADS * GSL;
    unsigned short* osb = osSwz + (size_t)b * HEADS * GSL * DHEAD;

    // kv = mean over heads of normalized slice tokens
    for (int i = tid; i < GSL * DHEAD; i += 256) {
        int g = i >> 6, c = i & 63;
        float s = 0.0f;
        for (int h = 0; h < HEADS; ++h)
            s += st[(h * GSL + g) * DHEAD + c]
               * __builtin_amdgcn_rcpf(nrm[h * GSL + g] + 1e-5f);
        kv[i] = s * (1.0f / HEADS);
    }
    __syncthreads();

    // k = kv @ Wk^T ; v = kv @ Wv^T
    for (int i = tid; i < GSL * DHEAD; i += 256) {
        int g = i >> 6, c = i & 63;
        float sk = 0.0f, sv = 0.0f;
        for (int j = 0; j < DHEAD; ++j) {
            float t = kv[g * DHEAD + j];
            sk += t * Wk[c * DHEAD + j];
            sv += t * Wv[c * DHEAD + j];
        }
        kn[i] = sk; vv[i] = sv;
    }
    __syncthreads();

    if (tid < GSL) {
        float s = 0.0f;
        for (int c = 0; c < DHEAD; ++c) { float t = kn[tid * DHEAD + c]; s += t * t; }
        rn[tid] = __builtin_amdgcn_rcpf(fmaxf(sqrtf(s), 1e-12f));
    }
    __syncthreads();
    for (int i = tid; i < GSL * DHEAD; i += 256) kn[i] *= rn[i >> 6];
    __syncthreads();

    const float srs = srs_p[0];
    for (int pair = tid; pair < HEADS * GSL; pair += 256) {
        int h = pair >> 6, g = pair & 63;
        float row[DHEAD], q[DHEAD], lg[GSL];
        float invn = __builtin_amdgcn_rcpf(nrm[h * GSL + g] + 1e-5f);
        for (int j = 0; j < DHEAD; ++j)
            row[j] = st[(h * GSL + g) * DHEAD + j] * invn;
        float qs = 0.0f;
        for (int c = 0; c < DHEAD; ++c) {
            float s = 0.0f;
            for (int j = 0; j < DHEAD; ++j) s += row[j] * Wq[c * DHEAD + j];
            q[c] = s; qs += s * s;
        }
        qs = __builtin_amdgcn_rcpf(fmaxf(sqrtf(qs), 1e-12f));
        float scale = attn_scale[h], mx = -1e30f;
        for (int j = 0; j < GSL; ++j) {
            float s = 0.0f;
            for (int c = 0; c < DHEAD; ++c) s += q[c] * kn[j * DHEAD + c];
            lg[j] = s * qs * scale;
            mx = fmaxf(mx, lg[j]);
        }
        float sum = 0.0f;
        for (int j = 0; j < GSL; ++j) { lg[j] = __expf(lg[j] - mx); sum += lg[j]; }
        float isum = __builtin_amdgcn_rcpf(sum);
        for (int c = 0; c < DHEAD; ++c) {
            float o = 0.0f;
            for (int j = 0; j < GSL; ++j) o += lg[j] * vv[j * DHEAD + c];
            o = o * isum + srs * row[c];
            // store in B-fragment swizzled layout: matrix [K=g=64][N=c=64]
            osb[(size_t)h * GSL * DHEAD + swzB(g, c, DHEAD)] = f2bf(o);
        }
    }
}

// ---------------------------------------------------------------------------
// Phase C: recompute slice weights per tile, scatter out_slice back to
// tokens (w @ os), then fused 512->256 output GEMM. 8 waves = 8 heads.
// ---------------------------------------------------------------------------
__global__ __launch_bounds__(256) void phaseC_kernel(
    const float* __restrict__ x, const float* __restrict__ sb,
    const unsigned short* __restrict__ WxsSwz,
    const float* __restrict__ bcomb, const float* __restrict__ temp,
    const unsigned short* __restrict__ osSwz,
    const unsigned short* __restrict__ WoutSwz,
    const float* __restrict__ bout, float* __restrict__ out)
{
    extern __shared__ unsigned char smem[];
    unsigned short* xs    = (unsigned short*)smem;                    // [TM][XS_LD]
    float*          sbls  = (float*)(smem + TM * XS_LD * 2);          // [TM][64]
    unsigned short* wls   = (unsigned short*)(sbls + TM * GSL);       // [8][TM][WL_LD]
    unsigned short* osls  = wls + HEADS * TM * WL_LD;                 // [8][4096] swz
    unsigned short* tmpls = osls + HEADS * GSL * DHEAD;               // [TM][TMP_LD]

    const int tid  = threadIdx.x;
    const int wave = tid >> 5;
    const int lane = tid & 31;
    const int batch = blockIdx.x / TILES;
    const int tile  = blockIdx.x % TILES;
    const int tok0  = tile * TM;

    if (tid == 0) { __builtin_prefetch(WoutSwz, 0, 1); }

    const float* xg = x + ((size_t)batch * NTOK + tok0) * DIMX;
    for (int i = tid; i < TM * DIMX / 4; i += 256) {
        float4 f = ((const float4*)xg)[i];
        int e = i * 4, row = e >> 8, col = e & 255;
        unsigned int* d = (unsigned int*)(xs + row * XS_LD + col);
        d[0] = f2bf2(f.x, f.y);
        d[1] = f2bf2(f.z, f.w);
    }
    const float* sbg = sb + ((size_t)batch * NTOK + tok0) * GSL;
    for (int i = tid; i < TM * GSL; i += 256) sbls[i] = sbg[i];
    const unsigned short* osg = osSwz + (size_t)batch * HEADS * GSL * DHEAD;
    for (int i = tid; i < HEADS * GSL * DHEAD; i += 256) osls[i] = osg[i];
    __syncthreads();

    const int h = wave;
    const float invT = __builtin_amdgcn_rcpf(temp[h]);
    unsigned short* wlsh = wls + h * TM * WL_LD;
    compute_slice_weights(xs, sbls, WxsSwz, bcomb, invT, wlsh, h, lane, nullptr);

    // GEMM1: tmp[tok][h*64+c] = w[tok][g] @ os[h][g][c]   (K = 64)
    {
        const unsigned short* oslsh = osls + h * GSL * DHEAD;
        const int n15 = lane & 15, half = lane >> 4;
#pragma unroll
        for (int r = 0; r < 2; ++r)
#pragma unroll
            for (int ct = 0; ct < 4; ++ct) {
                v8f acc = vzero8();
#pragma unroll
                for (int kk = 0; kk < 2; ++kk) {
                    v16bf a = ldA(wlsh, WL_LD, 16 * r, kk * 32, lane);
                    v16bf b = ldBswz(oslsh, DHEAD, kk, 16 * ct, lane);
                    acc = wmma_bf16(a, b, acc);
                }
#pragma unroll
                for (int v = 0; v < 8; ++v) {
                    int tok = 16 * r + v + 8 * half;
                    tmpls[tok * TMP_LD + h * 64 + 16 * ct + n15] = f2bf(acc[v]);
                }
            }
    }
    __syncthreads();

    // GEMM2: out[32][256] = tmp[32][512] @ W_out[512][256] + b_out
    {
        float* outg = out + ((size_t)batch * NTOK + tok0) * DIMX;
        const int n15 = lane & 15, half = lane >> 4;
#pragma unroll
        for (int t = 0; t < 4; ++t) {
            int tileid = wave + t * 8;           // 0..31
            int r = tileid >> 4, ct = tileid & 15;
            v8f acc = vzero8();
#pragma unroll
            for (int kk = 0; kk < 16; ++kk) {
                v16bf a = ldA(tmpls, TMP_LD, 16 * r, kk * 32, lane);
                v16bf b = ldBswz(WoutSwz, DIMX, kk, 16 * ct, lane);
                acc = wmma_bf16(a, b, acc);
            }
            float bb = bout[16 * ct + n15];
#pragma unroll
            for (int v = 0; v < 8; ++v) {
                int tok = 16 * r + v + 8 * half;
                outg[(size_t)tok * DIMX + 16 * ct + n15] = acc[v] + bb;
            }
        }
    }
}

// ---------------------------------------------------------------------------
// Host launcher
// ---------------------------------------------------------------------------
extern "C" void kernel_launch(void* const* d_in, const int* in_sizes, int n_in,
                              void* d_out, int out_size, void* d_ws, size_t ws_size,
                              hipStream_t stream) {
    (void)in_sizes; (void)n_in; (void)out_size; (void)ws_size;
    const float* x        = (const float*)d_in[0];
    const float* sb       = (const float*)d_in[1];
    const float* W_fx     = (const float*)d_in[2];
    const float* b_fx     = (const float*)d_in[3];
    const float* W_x      = (const float*)d_in[4];
    const float* b_x      = (const float*)d_in[5];
    const float* W_slice  = (const float*)d_in[6];
    const float* b_slice  = (const float*)d_in[7];
    const float* temp     = (const float*)d_in[8];
    const float* W_q      = (const float*)d_in[9];
    const float* W_k      = (const float*)d_in[10];
    const float* W_v      = (const float*)d_in[11];
    const float* srs      = (const float*)d_in[12];
    const float* ascale   = (const float*)d_in[13];
    const float* W_out    = (const float*)d_in[14];
    const float* b_out    = (const float*)d_in[15];
    float* out = (float*)d_out;

    // workspace layout (bytes)
    unsigned char* ws = (unsigned char*)d_ws;
    unsigned short* WfxSwz  = (unsigned short*)(ws + 0);        // 262144 B
    unsigned short* WxsSwz  = (unsigned short*)(ws + 262144);   // 262144 B
    unsigned short* WoutSwz = (unsigned short*)(ws + 524288);   // 262144 B
    float*          bcomb   = (float*)(ws + 786432);            // 2048 B
    float*          normA   = (float*)(ws + 788480);            // 8192 B
    float*          stA     = (float*)(ws + 796672);            // 524288 B
    unsigned short* osSwz   = (unsigned short*)(ws + 1320960);  // 262144 B

    const size_t smemA = (size_t)TM * XS_LD * 2 + (size_t)TM * GSL * 4
                       + (size_t)HEADS * TM * WL_LD * 2
                       + (size_t)HEADS * DHEAD * TM * 2;                    // 94720
    const size_t smemC = (size_t)TM * XS_LD * 2 + (size_t)TM * GSL * 4
                       + (size_t)HEADS * TM * WL_LD * 2
                       + (size_t)HEADS * GSL * DHEAD * 2
                       + (size_t)TM * TMP_LD * 2;                           // 160768

    prep_kernel<<<512, 256, 0, stream>>>(W_fx, W_x, W_slice, b_x, b_slice,
                                         W_out, WfxSwz, WxsSwz, WoutSwz, bcomb);
    zero_kernel<<<520, 256, 0, stream>>>(normA, 133120);  // norm + slice_token

    phaseA_kernel<<<BATCH * TILES, 256, smemA, stream>>>(
        x, sb, WxsSwz, WfxSwz, bcomb, b_fx, temp, stA, normA);

    phaseB_kernel<<<BATCH, 256, 0, stream>>>(stA, normA, W_q, W_k, W_v,
                                             srs, ascale, osSwz);

    phaseC_kernel<<<BATCH * TILES, 256, smemC, stream>>>(
        x, sb, WxsSwz, bcomb, temp, osSwz, WoutSwz, b_out, out);
}
